// Seq2Seq_RNN_RNN_Model_15169824489877
// MI455X (gfx1250) — matrix operand order repaired and checked by reading, not compiled, not run on hardware
//
#include <hip/hip_runtime.h>
#include <hip/hip_bf16.h>
#include <stdint.h>

#define VOCAB 12149
#define EMB 50
#define TAGS 9
#define HID 128
#define STARTTOK 7
#define BSZ 256
#define TLEN 512

typedef __attribute__((ext_vector_type(16))) _Float16 v16h;
typedef __attribute__((ext_vector_type(8)))  float    v8f;

#define WMMA_F16(Cacc, Afrag, Bfrag) \
  Cacc = __builtin_amdgcn_wmma_f32_16x16x32_f16(false, Afrag, false, Bfrag, (short)0, Cacc, false, false)

// Branch-free tanh on the serial critical path.
// Prefer native v_tanh_f32 (CDNA5 transcendental); fallback uses v_exp_f32 + v_rcp_f32
// with correct saturation at +/-1 via inf/0 (no EXEC divergence, no branches).
__device__ __forceinline__ float fast_tanh(float x) {
#if __has_builtin(__builtin_amdgcn_tanhf)
  return __builtin_amdgcn_tanhf(x);
#else
  // tanh(x) = 1 - 2 / (exp2(2*log2(e)*x) + 1)
  float e = __builtin_amdgcn_exp2f(x * 2.885390081777927f);
  return 1.0f - 2.0f * __builtin_amdgcn_rcpf(e + 1.0f);
#endif
}

// ---------------- Prep kernel 1: embW[v][c] = sum_e emb[v][e] * enc_Wx[e][c] ----------------
__global__ void embw_kernel(const float* __restrict__ emb, const float* __restrict__ Wx,
                            float* __restrict__ embW) {
  int v = blockIdx.x;
  int c = threadIdx.x;
  __shared__ float e[EMB];
  if (threadIdx.x < EMB) e[threadIdx.x] = emb[v * EMB + threadIdx.x];
  __syncthreads();
  float s = 0.f;
#pragma unroll 5
  for (int k = 0; k < EMB; ++k) s += e[k] * Wx[k * HID + c];
  embW[v * HID + c] = s;
}

// ---------------- Prep kernel 2: decXW[r][c] = sum_e dec_emb[r][e] * dec_Wx[e][c], rows padded to 16 --------
__global__ void decxw_kernel(const float* __restrict__ demb, const float* __restrict__ Wx,
                             float* __restrict__ decXW) {
  int c = threadIdx.x;
  for (int r = 0; r < 16; ++r) {
    float s = 0.f;
    if (r < TAGS) {
      for (int k = 0; k < EMB; ++k) s += demb[r * EMB + k] * Wx[k * HID + c];
    }
    decXW[r * HID + c] = s;
  }
}

// ---------------- Prep kernel 3: pack a [128 x ncols] f32 matrix into WMMA B-fragment f16 layout ----------
// Fragment (kb, w): lane l, dword v  <->  k = kb*32 + (l>>4)*16 + 2v (+half), n = 16*w + (l&15)
// dst dword index = ((kb*nw + w)*32 + lane)*8 + v   (per-lane 32B contiguous)
__global__ void pack_b_kernel(const float* __restrict__ M, int ncols, int nw,
                              uint32_t* __restrict__ dst) {
  int idx = blockIdx.x * blockDim.x + threadIdx.x;
  int total = 4 * nw * 32 * 8;
  if (idx >= total) return;
  int v = idx & 7;
  int lane = (idx >> 3) & 31;
  int rest = idx >> 8;
  int w = rest % nw;
  int kb = rest / nw;
  int k0 = kb * 32 + ((lane >> 4) << 4) + 2 * v;
  int n = 16 * w + (lane & 15);
  float f0 = (n < ncols) ? M[k0 * ncols + n] : 0.f;
  float f1 = (n < ncols) ? M[(k0 + 1) * ncols + n] : 0.f;
  union { _Float16 h[2]; uint32_t u; } p;
  p.h[0] = (_Float16)f0;
  p.h[1] = (_Float16)f1;
  dst[idx] = p.u;
}

// ---------------- Main recurrent kernel ----------------
// 16 workgroups (one per 16-row batch stripe) x 9 waves.
// Waves 0..7: col-tile w of h via 4x v_wmma_f32_16x16x32_f16 per step (K=128).
// Wave 8: overlapped output projection h @ W + b (K=128, N padded to 16).
__global__ __launch_bounds__(288) void seq2seq_kernel(
    const int* __restrict__ inputs, const int* __restrict__ seq_label,
    const float* __restrict__ enc_b, const float* __restrict__ dec_b,
    const float* __restrict__ outb_g,
    const float* __restrict__ embW, const float* __restrict__ decXW,
    const uint32_t* __restrict__ encWhP, const uint32_t* __restrict__ decWhP,
    const uint32_t* __restrict__ WP,
    float* __restrict__ out) {
  const int lane = threadIdx.x & 31;
  const int wave = threadIdx.x >> 5;       // 0..8
  const int lane15 = lane & 15;
  const int lg = lane >> 4;                // 0/1
  const int b0 = blockIdx.x * 16;

  // h in A-fragment f16 layout, double buffered: [buf][ (kb*32 + lane)*8 + v ] dwords
  __shared__ __align__(32) uint32_t hA[2][4 * 32 * 8];
  __shared__ unsigned short encTok[16 * TLEN];
  __shared__ unsigned short decTok[16 * TLEN];

  // Stage tokens into LDS (u16) + zero h_{-1} buffer
  for (int i = threadIdx.x; i < 16 * TLEN; i += blockDim.x) {
    int m = i >> 9;
    int t = i & (TLEN - 1);
    encTok[i] = (unsigned short)inputs[(b0 + m) * TLEN + t];
    decTok[i] = (unsigned short)((t == 0) ? STARTTOK : seq_label[(b0 + m) * TLEN + t - 1]);
  }
  for (int i = threadIdx.x; i < 4 * 32 * 8; i += blockDim.x) hA[0][i] = 0u;

  // Resident B fragments (kept in VGPRs the whole kernel)
  v16h Benc[4], Bdec[4], Bw[4];
  float encb = 0.f, decb = 0.f, ob = 0.f;
  const int col = 16 * wave + lane15;      // meaningful for wave<8
  if (wave < 8) {
#pragma unroll
    for (int kb = 0; kb < 4; ++kb) {
      Benc[kb] = *(const v16h*)(encWhP + ((kb * 8 + wave) * 32 + lane) * 8);
      Bdec[kb] = *(const v16h*)(decWhP + ((kb * 8 + wave) * 32 + lane) * 8);
    }
    encb = enc_b[col];
    decb = dec_b[col];
  } else {
#pragma unroll
    for (int kb = 0; kb < 4; ++kb)
      Bw[kb] = *(const v16h*)(WP + ((kb * 1 + 0) * 32 + lane) * 8);
    ob = (lane15 < TAGS) ? outb_g[lane15] : 0.f;
  }

  // h-store address precompute (element (m, col) -> A-fragment slot in next buffer)
  const int kk = col & 31;
  const int kbst = col >> 5;
  const int grp = (kk >> 4) & 1;
  const int vr = (kk & 15) >> 1;
  const int hlf = kk & 1;

  __syncthreads();

  float x[8];
  // ---------------- Encoder: prefetch x for t=0 ----------------
  if (wave < 8) {
#pragma unroll
    for (int r = 0; r < 8; ++r) {
      int m = r + 8 * lg;
      int tok = (int)encTok[m * TLEN + 0];
      x[r] = embW[tok * HID + col];
    }
  }

  for (int t = 0; t < TLEN; ++t) {
    float xn[8];
    if (wave < 8) {
      // prefetch gather for t+1 (hides L2 latency behind WMMA chain)
      int tp = (t + 1 < TLEN) ? t + 1 : t;
#pragma unroll
      for (int r = 0; r < 8; ++r) {
        int m = r + 8 * lg;
        int tok = (int)encTok[m * TLEN + tp];
        xn[r] = embW[tok * HID + col];
      }
      // load A fragments of h_{t-1}
      const uint32_t* ab = &hA[t & 1][0];
      v16h A0 = *(const v16h*)(ab + (0 * 32 + lane) * 8);
      v16h A1 = *(const v16h*)(ab + (1 * 32 + lane) * 8);
      v16h A2 = *(const v16h*)(ab + (2 * 32 + lane) * 8);
      v16h A3 = *(const v16h*)(ab + (3 * 32 + lane) * 8);
      v8f c;
#pragma unroll
      for (int r = 0; r < 8; ++r) c[r] = x[r] + encb;
      WMMA_F16(c, A0, Benc[0]);
      WMMA_F16(c, A1, Benc[1]);
      WMMA_F16(c, A2, Benc[2]);
      WMMA_F16(c, A3, Benc[3]);
      // tanh + scatter into next A-fragment buffer
      _Float16* hh = (_Float16*)(&hA[(t + 1) & 1][0]);
#pragma unroll
      for (int r = 0; r < 8; ++r) {
        int m = r + 8 * lg;
        int dlane = m + 16 * grp;
        hh[((kbst * 32 + dlane) * 8 + vr) * 2 + hlf] = (_Float16)fast_tanh(c[r]);
      }
    }
    __syncthreads();
    if (wave < 8) {
#pragma unroll
      for (int r = 0; r < 8; ++r) x[r] = xn[r];
    }
  }

  // ---------------- Decoder (encoder final state sits in buf 0) ----------------
  if (wave < 8) {
#pragma unroll
    for (int r = 0; r < 8; ++r) {
      int m = r + 8 * lg;
      int tok = (int)decTok[m * TLEN + 0];
      x[r] = decXW[tok * HID + col];
    }
  }

  for (int t = 0; t < TLEN; ++t) {
    float xn[8];
    if (wave < 8) {
      int tp = (t + 1 < TLEN) ? t + 1 : t;
#pragma unroll
      for (int r = 0; r < 8; ++r) {
        int m = r + 8 * lg;
        int tok = (int)decTok[m * TLEN + tp];
        xn[r] = decXW[tok * HID + col];
      }
      const uint32_t* ab = &hA[t & 1][0];
      v16h A0 = *(const v16h*)(ab + (0 * 32 + lane) * 8);
      v16h A1 = *(const v16h*)(ab + (1 * 32 + lane) * 8);
      v16h A2 = *(const v16h*)(ab + (2 * 32 + lane) * 8);
      v16h A3 = *(const v16h*)(ab + (3 * 32 + lane) * 8);
      v8f c;
#pragma unroll
      for (int r = 0; r < 8; ++r) c[r] = x[r] + decb;
      WMMA_F16(c, A0, Bdec[0]);
      WMMA_F16(c, A1, Bdec[1]);
      WMMA_F16(c, A2, Bdec[2]);
      WMMA_F16(c, A3, Bdec[3]);
      _Float16* hh = (_Float16*)(&hA[(t + 1) & 1][0]);
#pragma unroll
      for (int r = 0; r < 8; ++r) {
        int m = r + 8 * lg;
        int dlane = m + 16 * grp;
        hh[((kbst * 32 + dlane) * 8 + vr) * 2 + hlf] = (_Float16)fast_tanh(c[r]);
      }
    } else if (t > 0) {
      // Wave 8: project h_{t-1} (same buffer the h-waves are reading) -> overlaps step t
      const uint32_t* ab = &hA[t & 1][0];
      v16h A0 = *(const v16h*)(ab + (0 * 32 + lane) * 8);
      v16h A1 = *(const v16h*)(ab + (1 * 32 + lane) * 8);
      v16h A2 = *(const v16h*)(ab + (2 * 32 + lane) * 8);
      v16h A3 = *(const v16h*)(ab + (3 * 32 + lane) * 8);
      v8f o;
#pragma unroll
      for (int r = 0; r < 8; ++r) o[r] = ob;
      WMMA_F16(o, A0, Bw[0]);
      WMMA_F16(o, A1, Bw[1]);
      WMMA_F16(o, A2, Bw[2]);
      WMMA_F16(o, A3, Bw[3]);
      int tprev = t - 1;
      if (lane15 < TAGS) {
#pragma unroll
        for (int r = 0; r < 8; ++r) {
          int m = r + 8 * lg;
          out[((size_t)(b0 + m) * TLEN + tprev) * TAGS + lane15] = o[r];
        }
      }
    }
    __syncthreads();
    if (wave < 8) {
#pragma unroll
      for (int r = 0; r < 8; ++r) x[r] = xn[r];
    }
  }

  // Epilogue: project final decoder state h_{TLEN-1} (in buf 0 since TLEN is even)
  if (wave == 8) {
    const uint32_t* ab = &hA[TLEN & 1][0];
    v16h A0 = *(const v16h*)(ab + (0 * 32 + lane) * 8);
    v16h A1 = *(const v16h*)(ab + (1 * 32 + lane) * 8);
    v16h A2 = *(const v16h*)(ab + (2 * 32 + lane) * 8);
    v16h A3 = *(const v16h*)(ab + (3 * 32 + lane) * 8);
    v8f o;
#pragma unroll
    for (int r = 0; r < 8; ++r) o[r] = ob;
    WMMA_F16(o, A0, Bw[0]);
    WMMA_F16(o, A1, Bw[1]);
    WMMA_F16(o, A2, Bw[2]);
    WMMA_F16(o, A3, Bw[3]);
    if (lane15 < TAGS) {
#pragma unroll
      for (int r = 0; r < 8; ++r) {
        int m = r + 8 * lg;
        out[((size_t)(b0 + m) * TLEN + (TLEN - 1)) * TAGS + lane15] = o[r];
      }
    }
  }
}

extern "C" void kernel_launch(void* const* d_in, const int* in_sizes, int n_in,
                              void* d_out, int out_size, void* d_ws, size_t ws_size,
                              hipStream_t stream) {
  (void)in_sizes; (void)n_in; (void)out_size; (void)ws_size;
  const int*   inputs    = (const int*)d_in[0];
  const int*   seq_label = (const int*)d_in[1];
  const float* emb       = (const float*)d_in[2];
  const float* enc_Wx    = (const float*)d_in[3];
  const float* enc_Wh    = (const float*)d_in[4];
  const float* enc_b     = (const float*)d_in[5];
  const float* dec_emb   = (const float*)d_in[6];
  const float* dec_Wx    = (const float*)d_in[7];
  const float* dec_Wh    = (const float*)d_in[8];
  const float* dec_b     = (const float*)d_in[9];
  const float* W         = (const float*)d_in[10];
  const float* b         = (const float*)d_in[11];
  float* out = (float*)d_out;

  uint8_t* ws = (uint8_t*)d_ws;
  // workspace layout (all 256B-aligned)
  float*    embW   = (float*)(ws);                                    // VOCAB*HID f32
  float*    decXW  = (float*)(ws + (size_t)VOCAB * HID * 4);          // 16*HID f32
  uint32_t* encWhP = (uint32_t*)(ws + (size_t)VOCAB * HID * 4 + 16 * HID * 4);
  uint32_t* decWhP = encWhP + 4 * 8 * 32 * 8;
  uint32_t* WP     = decWhP + 4 * 8 * 32 * 8;

  embw_kernel<<<VOCAB, 128, 0, stream>>>(emb, enc_Wx, embW);
  decxw_kernel<<<1, 128, 0, stream>>>(dec_emb, dec_Wx, decXW);
  pack_b_kernel<<<(4 * 8 * 32 * 8 + 255) / 256, 256, 0, stream>>>(enc_Wh, HID, 8, encWhP);
  pack_b_kernel<<<(4 * 8 * 32 * 8 + 255) / 256, 256, 0, stream>>>(dec_Wh, HID, 8, decWhP);
  pack_b_kernel<<<(4 * 1 * 32 * 8 + 255) / 256, 256, 0, stream>>>(W, TAGS, 1, WP);

  seq2seq_kernel<<<BSZ / 16, 288, 0, stream>>>(inputs, seq_label, enc_b, dec_b, b,
                                               embW, decXW, encWhP, decWhP, WP, out);
}
